// SelfAttention_74251394613999
// MI455X (gfx1250) — matrix operand order repaired and checked by reading, not compiled, not run on hardware
//
#include <hip/hip_runtime.h>

// ---------------------------------------------------------------------------
// Self-attention (SAGAN-style) for MI455X / gfx1250, wave32 + WMMA f16.
//   B=4, C=512, N=64*64=4096, d_qk=256, d_v=512
// Kernel 1: fused f/g/h 1x1-conv projections via v_wmma_f32_16x16x32_f16
// Kernel 2: flash attention (online softmax, O accum in VGPRs),
//           K tiles double-buffered via GLOBAL_LOAD_ASYNC_TO_LDS_B128.
// ---------------------------------------------------------------------------

typedef _Float16 v16h __attribute__((ext_vector_type(16)));
typedef float    v8f  __attribute__((ext_vector_type(8)));
typedef unsigned int u32x4 __attribute__((ext_vector_type(4)));
typedef float    f32x4 __attribute__((ext_vector_type(4)));

#define NTOK 4096
#define CC   512
#define DK   256
#define DV   512

// LDS strides (in halves / floats) chosen for conflict-free b128 frag loads
#define QKS 264   // 528B row stride -> bank step 132%64=4 dwords
#define VTS 72    // 144B row stride -> step 36 dwords
#define SSs 68    // f32 stride
#define PSs 72

// ---- gfx1250 async global->LDS path (probe-guarded, sync fallback) --------
#if defined(__AMDGCN__) && __has_builtin(__builtin_amdgcn_global_load_async_to_lds_b128) && \
    __has_builtin(__builtin_amdgcn_s_wait_asynccnt)
#define HAVE_ASYNC 1
#else
#define HAVE_ASYNC 0
#endif

// builtin signature (leaked by clang diagnostic): param0 is
//   'int __vector(4) __device__ *'  i.e.  v4i addrspace(1)*
typedef int v4i __attribute__((vector_size(16)));
typedef __attribute__((address_space(1))) v4i gv4i;   // global (__device__)
typedef __attribute__((address_space(3))) v4i lv4i;   // LDS (__shared__)

static __device__ __forceinline__ void async_copy_b128(const _Float16* g,
                                                       _Float16* l) {
#if HAVE_ASYNC
  // AS pointers materialized via integer casts: generic global ptr bit
  // pattern == AS1; generic LDS ptr low 32 bits == LDS offset (ISA 10.2).
  __builtin_amdgcn_global_load_async_to_lds_b128(
      (gv4i*)(size_t)g, (lv4i*)(unsigned int)(size_t)l, 0, 0);
#else
  *(u32x4*)l = *(const u32x4*)g;
#endif
}

#if HAVE_ASYNC
#define WAIT_ASYNC(n) __builtin_amdgcn_s_wait_asynccnt(n)
#else
#define WAIT_ASYNC(n) ((void)0)
#endif

static __device__ __forceinline__ v8f wmma_f16(v16h a, v16h b, v8f c) {
  return __builtin_amdgcn_wmma_f32_16x16x32_f16(false, a, false, b, (short)0, c,
                                                false, false);
}

union Frag { v16h h; u32x4 q[2]; };
union Pack8 { u32x4 q; _Float16 h[8]; };

// ---------------------------------------------------------------------------
// Kernel 1: Q/K/V projections.  out[n,o] = sum_c x[b,c,n] * W[o,c] + bias[o]
// grid = (N/64, 1024/64, B); combined o-space: [0,256)=f, [256,512)=g, rest=h
// ---------------------------------------------------------------------------
__global__ __launch_bounds__(256, 1)
void proj_kernel(const float* __restrict__ x,
                 const float* __restrict__ fw, const float* __restrict__ fb,
                 const float* __restrict__ gw, const float* __restrict__ gb,
                 const float* __restrict__ hw, const float* __restrict__ hb,
                 _Float16* __restrict__ Qg, _Float16* __restrict__ Kg,
                 _Float16* __restrict__ Vg) {
  const int XS = 40, WS = 40;                 // 32 k + 8 pad halves
  __shared__ _Float16 xt[64 * 40];            // A tile: [n][c]  (f16)
  __shared__ _Float16 wl[64 * 40];            // B tile: [o][c]  (f16)

  const int t = threadIdx.x;
  const int wave = t >> 5, lane = t & 31, hi = lane >> 4, ln = lane & 15;
  const int b = blockIdx.z, n0 = blockIdx.x * 64, o0 = blockIdx.y * 64;

  const float* Wm; const float* bia; _Float16* dst; int ow, D;
  if (o0 < 256)      { Wm = fw; bia = fb; dst = Qg; ow = o0;       D = DK; }
  else if (o0 < 512) { Wm = gw; bia = gb; dst = Kg; ow = o0 - 256; D = DK; }
  else               { Wm = hw; bia = hb; dst = Vg; ow = o0 - 512; D = DV; }

  const v8f vzero = {0, 0, 0, 0, 0, 0, 0, 0};
  v8f acc[2]; acc[0] = vzero; acc[1] = vzero;

  for (int c0 = 0; c0 < CC; c0 += 32) {
    __syncthreads();
    {
      // x tile: 32c x 64n fp32 -> f16, transposed into xt[n][c]
      int ci = t >> 3, noff = (t & 7) << 3;
      const float* src = x + ((size_t)b * CC + c0 + ci) * NTOK + n0 + noff;
      f32x4 a0 = *(const f32x4*)(src);
      f32x4 a1 = *(const f32x4*)(src + 4);
#pragma unroll
      for (int q = 0; q < 4; ++q) xt[(noff + q) * XS + ci]     = (_Float16)a0[q];
#pragma unroll
      for (int q = 0; q < 4; ++q) xt[(noff + 4 + q) * XS + ci] = (_Float16)a1[q];

      // W tile: 64o x 32c fp32 -> f16, kept row-major (already [o][c])
      int oi = t >> 2, coff = (t & 3) << 3;
      const float* wsrc = Wm + (size_t)(ow + oi) * CC + c0 + coff;
      f32x4 w0 = *(const f32x4*)(wsrc);
      f32x4 w1 = *(const f32x4*)(wsrc + 4);
      Pack8 pk;
#pragma unroll
      for (int q = 0; q < 4; ++q) pk.h[q]     = (_Float16)w0[q];
#pragma unroll
      for (int q = 0; q < 4; ++q) pk.h[4 + q] = (_Float16)w1[q];
      *(u32x4*)(wl + oi * WS + coff) = pk.q;
    }
    __syncthreads();

#pragma unroll
    for (int tt = 0; tt < 2; ++tt) {          // 2 of 16 output tiles per wave
      int tile = wave * 2 + tt, ng = tile >> 2, og = tile & 3;
      Frag af, bf;
      int arow = ng * 16 + ln;
      af.q[0] = *(const u32x4*)(xt + arow * XS + (hi << 3));
      af.q[1] = *(const u32x4*)(xt + arow * XS + (hi << 3) + 16);
      int brow = og * 16 + ln;
      bf.q[0] = *(const u32x4*)(wl + brow * WS + (hi << 3));
      bf.q[1] = *(const u32x4*)(wl + brow * WS + (hi << 3) + 16);
      acc[tt] = wmma_f16(af.h, bf.h, acc[tt]);
    }
  }

#pragma unroll
  for (int tt = 0; tt < 2; ++tt) {
    int tile = wave * 2 + tt, ng = tile >> 2, og = tile & 3;
    int o_loc = og * 16 + ln;
    float bv = bia[ow + o_loc];
#pragma unroll
    for (int vr = 0; vr < 8; ++vr) {
      int n_loc = ng * 16 + (hi << 3) + vr;   // C/D layout: row = vr + hi*8
      dst[((size_t)b * NTOK + n0 + n_loc) * D + ow + o_loc] =
          (_Float16)(acc[tt][vr] + bv);
    }
  }
}

// ---------------------------------------------------------------------------
// Kernel 2: flash attention.  grid = (N/64, B), 8 waves / block.
//   S = Q K^T (d=256), online softmax over j, O += P V (d_v=512)
// Wave w owns query row-group rg=w&3 and c-half cgb=(w>>2)*16 (16 tiles of O).
// K tiles double-buffered with async global->LDS loads.
// ---------------------------------------------------------------------------
__global__ __launch_bounds__(256, 1)
void attn_kernel(const _Float16* __restrict__ Qg,
                 const _Float16* __restrict__ Kg,
                 const _Float16* __restrict__ Vg,
                 float* __restrict__ out) {
  extern __shared__ char smem_raw[];
  _Float16* Qs  = (_Float16*)smem_raw;          // [64][QKS]
  _Float16* Ks0 = Qs + 64 * QKS;                // [2][64][QKS] double buffer
  _Float16* Ks1 = Ks0 + 64 * QKS;
  _Float16* Vt  = Ks1 + 64 * QKS;               // [512][VTS]  V transposed [c][j]
  float*    Ssm = (float*)(Vt + 512 * VTS);     // [64][SSs]   raw scores
  _Float16* Ps  = (_Float16*)(Ssm + 64 * SSs);  // [64][PSs]   exp probs (f16)
  float* mrow = (float*)(Ps + 64 * PSs);
  float* lrow = mrow + 64;
  float* srow = lrow + 64;

  const int t = threadIdx.x;
  const int wave = t >> 5, lane = t & 31, hi = lane >> 4, ln = lane & 15;
  const int b = blockIdx.y, i0 = blockIdx.x * 64;
  const int rg = wave & 3;                 // query row-group (16 rows)
  const int cgb = (wave >> 2) << 4;        // v-column group base (16 tiles)

  // ---- async-load Q tile (64 x 256 f16) and K tile 0 into LDS ----
  const _Float16* Qbase = Qg + ((size_t)b * NTOK + i0) * DK;
  const _Float16* Kall  = Kg + (size_t)b * NTOK * DK;
#pragma unroll
  for (int r = 0; r < 8; ++r) {
    int e = t + r * 256;                   // 2048 chunks of 8 halves
    int row = e >> 5, off = (e & 31) << 3;
    async_copy_b128(Qbase + row * DK + off, Qs + row * QKS + off);
  }
#pragma unroll
  for (int r = 0; r < 8; ++r) {
    int e = t + r * 256;
    int row = e >> 5, off = (e & 31) << 3;
    async_copy_b128(Kall + row * DK + off, Ks0 + row * QKS + off);
  }

  const v8f vzero = {0, 0, 0, 0, 0, 0, 0, 0};
  v8f oacc[16];
#pragma unroll
  for (int i = 0; i < 16; ++i) oacc[i] = vzero;

  for (int jt = 0; jt < NTOK / 64; ++jt) {
    _Float16* Kcur = (jt & 1) ? Ks1 : Ks0;
    _Float16* Knxt = (jt & 1) ? Ks0 : Ks1;
    __syncthreads();                       // all waves done with Knxt & Vt/Ps

    // issue async loads for the NEXT K tile (stays in flight behind compute)
    if (jt + 1 < NTOK / 64) {
      const _Float16* Kb = Kall + (size_t)(jt + 1) * 64 * DK;
#pragma unroll
      for (int r = 0; r < 8; ++r) {
        int e = t + r * 256;
        int row = e >> 5, off = (e & 31) << 3;
        async_copy_b128(Kb + row * DK + off, Knxt + row * QKS + off);
      }
    }

    // V tile 64x512, transposed into Vt[c][j] (needs VGPR round-trip)
    const _Float16* Vbase = Vg + ((size_t)b * NTOK + jt * 64) * DV;
#pragma unroll
    for (int r = 0; r < 16; ++r) {
      int e = t + r * 256;
      int row = e >> 6, off = (e & 63) << 3;
      Pack8 u;
      u.q = *(const u32x4*)(Vbase + row * DV + off);
#pragma unroll
      for (int q = 0; q < 8; ++q) Vt[(off + q) * VTS + row] = u.h[q];
    }
    if (jt + 1 < NTOK / 64)                // pull next V tile toward L2
      __builtin_prefetch(Vbase + 64 * DV + t * 128, 0, 1);

    // current K tile (and Q on jt==0) must be resident; next tile may fly
    if (jt + 1 < NTOK / 64) WAIT_ASYNC(8); else WAIT_ASYNC(0);
    __syncthreads();

    // ---- S = Q K^T : 16 tiles of 16x16, 2 per wave, K=256 in 8 steps ----
#pragma unroll
    for (int tt = 0; tt < 2; ++tt) {
      int tile = wave * 2 + tt, rgs = tile >> 2, cgs = tile & 3;
      v8f sacc = vzero;
#pragma unroll
      for (int ks = 0; ks < 8; ++ks) {
        Frag af, bf;
        int arow = rgs * 16 + ln;
        int akb = ks * 32 + (hi << 3);
        af.q[0] = *(const u32x4*)(Qs + arow * QKS + akb);
        af.q[1] = *(const u32x4*)(Qs + arow * QKS + akb + 16);
        int brow = cgs * 16 + ln;            // key row j = B column
        int bkb = ks * 32 + (hi << 3);
        bf.q[0] = *(const u32x4*)(Kcur + brow * QKS + bkb);
        bf.q[1] = *(const u32x4*)(Kcur + brow * QKS + bkb + 16);
        sacc = wmma_f16(af.h, bf.h, sacc);
      }
#pragma unroll
      for (int vr = 0; vr < 8; ++vr)
        Ssm[(rgs * 16 + (hi << 3) + vr) * SSs + cgs * 16 + ln] = sacc[vr];
    }
    __syncthreads();

    // ---- online softmax, one thread per query row ----
    if (t < 64) {
      float mold, lold;
      if (jt == 0) { mold = -1e30f; lold = 0.0f; }
      else         { mold = mrow[t]; lold = lrow[t]; }
      float mx = mold;
#pragma unroll 8
      for (int j = 0; j < 64; ++j) mx = fmaxf(mx, Ssm[t * SSs + j]);
      float sc = __expf(mold - mx);
      float l = lold * sc;
#pragma unroll 8
      for (int j = 0; j < 64; ++j) {
        float e = __expf(Ssm[t * SSs + j] - mx);
        Ps[t * PSs + j] = (_Float16)e;
        l += e;
      }
      mrow[t] = mx; lrow[t] = l; srow[t] = sc;
    }
    __syncthreads();

    // ---- rescale O accumulators, then O += P * V ----
    float scv[8];
#pragma unroll
    for (int vr = 0; vr < 8; ++vr) scv[vr] = srow[rg * 16 + (hi << 3) + vr];
#pragma unroll
    for (int tc = 0; tc < 16; ++tc)
#pragma unroll
      for (int vr = 0; vr < 8; ++vr) oacc[tc][vr] *= scv[vr];

#pragma unroll
    for (int kj = 0; kj < 2; ++kj) {       // j-dim 64 in two K=32 steps
      Frag pa;
      int arow = rg * 16 + ln;
      int akb = kj * 32 + (hi << 3);
      pa.q[0] = *(const u32x4*)(Ps + arow * PSs + akb);
      pa.q[1] = *(const u32x4*)(Ps + arow * PSs + akb + 16);
#pragma unroll
      for (int tc = 0; tc < 16; ++tc) {    // A-frag reused across 16 c-tiles
        Frag vb;
        int brow = (cgb + tc) * 16 + ln;   // c index = B column
        int bkb = kj * 32 + (hi << 3);
        vb.q[0] = *(const u32x4*)(Vt + brow * VTS + bkb);
        vb.q[1] = *(const u32x4*)(Vt + brow * VTS + bkb + 16);
        oacc[tc] = wmma_f16(pa.h, vb.h, oacc[tc]);
      }
    }
  }

  // ---- epilogue: normalize, transpose 16x16 tiles via per-wave LDS
  //      staging so stores are contiguous along i (2x b128 per lane) ----
  float invl[8];
#pragma unroll
  for (int vr = 0; vr < 8; ++vr)
    invl[vr] = 1.0f / lrow[rg * 16 + (hi << 3) + vr];

  const int STG = 20;                      // 80B row stride, bank-clean
  float* stg = Ssm + wave * (16 * STG);    // per-wave region, no barrier needed
  const int cl = lane >> 1, ib = (lane & 1) << 3;
#pragma unroll
  for (int tc = 0; tc < 16; ++tc) {
    // write transposed: stg[c_local][i_local]
#pragma unroll
    for (int vr = 0; vr < 8; ++vr)
      stg[ln * STG + (hi << 3) + vr] = oacc[tc][vr] * invl[vr];
    // wave-synchronous readback: lane -> (c_local=lane/2, i half)
    f32x4 v0 = *(const f32x4*)(stg + cl * STG + ib);
    f32x4 v1 = *(const f32x4*)(stg + cl * STG + ib + 4);
    int c = (cgb + tc) * 16 + cl;
    float* op = out + ((size_t)b * CC + c) * NTOK + i0 + rg * 16 + ib;
    *(f32x4*)op = v0;
    *(f32x4*)(op + 4) = v1;
  }
}

// ---------------------------------------------------------------------------
// Host launcher
// ---------------------------------------------------------------------------
extern "C" void kernel_launch(void* const* d_in, const int* in_sizes, int n_in,
                              void* d_out, int out_size, void* d_ws,
                              size_t ws_size, hipStream_t stream) {
  (void)in_sizes; (void)n_in; (void)out_size; (void)ws_size;
  const float* x  = (const float*)d_in[0];
  const float* fw = (const float*)d_in[1];
  const float* fb = (const float*)d_in[2];
  const float* gw = (const float*)d_in[3];
  const float* gb = (const float*)d_in[4];
  const float* hw = (const float*)d_in[5];
  const float* hb = (const float*)d_in[6];
  float* out = (float*)d_out;

  _Float16* Qg = (_Float16*)d_ws;                     //  8 MB
  _Float16* Kg = Qg + (size_t)4 * NTOK * DK;          //  8 MB
  _Float16* Vg = Kg + (size_t)4 * NTOK * DK;          // 16 MB

  proj_kernel<<<dim3(NTOK / 64, 1024 / 64, 4), 256, 0, stream>>>(
      x, fw, fb, gw, gb, hw, hb, Qg, Kg, Vg);

  constexpr size_t SMEM_BYTES =
      (size_t)(3 * 64 * QKS + 512 * VTS + 64 * PSs) * 2  // f16 regions
      + (size_t)(64 * SSs + 3 * 64) * 4;                 // f32 regions
  attn_kernel<<<dim3(NTOK / 64, 4), 256, SMEM_BYTES, stream>>>(Qg, Kg, Vg, out);
}
